// TransformerBlockHybrid_65481071398224
// MI455X (gfx1250) — compile-verified, matrix operand order checked by reading, not run on hardware
//
#include <hip/hip_runtime.h>
#include <hip/hip_bf16.h>

// ---------------------------------------------------------------------------
// TransformerBlockHybrid for MI455X (gfx1250, wave32, WMMA).
// B=4, S=1024, E=1024, H=16, D=64, FFN=4096. All GEMMs are bf16 NT-GEMMs
// (C = A*B^T, A:[M,K] row-major, B:[N,K] row-major) via
// v_wmma_f32_16x16x32_bf16 with f32 accumulation, staged through LDS with
// gfx1250 async copies (global_load_async_to_lds_b128, ASYNCcnt) and
// double-buffered tiles so tile k+1 loads overlap tile k WMMA math.
// ---------------------------------------------------------------------------

#define B_   4
#define S_   1024
#define E_   1024
#define H_   16
#define D_   64
#define FFN_ 4096

#define USE_ASYNC_COPY 1

typedef unsigned short ushort_t;
typedef __attribute__((ext_vector_type(16))) __bf16 v16bf;
typedef __attribute__((ext_vector_type(8)))  float  v8f;

struct BFrag { uint4 lo; uint4 hi; };
static_assert(sizeof(BFrag) == 32, "frag size");

__device__ __forceinline__ ushort_t f2bf(float f) {
    unsigned int u = __float_as_uint(f);
    u += 0x7FFFu + ((u >> 16) & 1u);   // round-to-nearest-even
    return (ushort_t)(u >> 16);
}
__device__ __forceinline__ float bf2f(ushort_t h) {
    return __uint_as_float(((unsigned int)h) << 16);
}

// gfx1250 async copy: per-lane 16B global -> LDS, tracked by ASYNCcnt.
__device__ __forceinline__ void async_load_b128(unsigned lds_addr, const void* gptr) {
#if USE_ASYNC_COPY
    asm volatile("global_load_async_to_lds_b128 %0, %1, off"
                 :: "v"(lds_addr), "v"(gptr)
                 : "memory");
#endif
}
__device__ __forceinline__ void async_wait0() {
#if USE_ASYNC_COPY
#if __has_builtin(__builtin_amdgcn_s_wait_asynccnt)
    __builtin_amdgcn_s_wait_asynccnt(0);
#else
    asm volatile("s_wait_asynccnt 0x0" ::: "memory");
#endif
#endif
}

// Low 32 bits of a generic LDS pointer == wave-relative LDS byte address
// (flat->LDS mapping truncates to addr[31:0], ISA 10.2).
__device__ __forceinline__ unsigned lds_byte_addr(const void* p) {
    return (unsigned)(unsigned long long)p;
}

// Load one 16x32 bf16 operand fragment from a row-major LDS tile.
// ISA 7.12.2 (16-bit A 16x32): lane l -> row (l&15); with h = l>>4 the lane
// holds K = [h*8, h*8+8) in VGPR0-3 and K = [16+h*8, 16+h*8+8) in VGPR4-7,
// i.e. two contiguous 16-byte chunks. B operand symmetric (lane = column).
__device__ __forceinline__ v16bf load_frag(const ushort_t* base, int stride, int lane) {
    int r = lane & 15;
    int h = lane >> 4;
    const ushort_t* p = base + r * stride + h * 8;
    BFrag f;
    f.lo = *reinterpret_cast<const uint4*>(p);
    f.hi = *reinterpret_cast<const uint4*>(p + 16);
    return __builtin_bit_cast(v16bf, f);
}

// ---------------------------------------------------------------------------
// Batched NT GEMM: C[z] = act(scale * (A[z]*B^T[z] (+ A2[z]*B2^T[z])) + bias)
// Requires M % BM == 0, N % BN == 0, K % 64 == 0 (true for all shapes here).
// ---------------------------------------------------------------------------
template<int BM, int BN, int WM, int WN, bool OUT_BF16, bool RELU>
__global__ __launch_bounds__(256)
void gemm_nt_kernel(const ushort_t* __restrict__ A,  int lda, long long strideA,
                    const ushort_t* __restrict__ Bm, int ldb, long long strideB,
                    void* __restrict__ Cp,           int ldc, long long strideC,
                    const float* __restrict__ bias,  float scale,
                    const ushort_t* __restrict__ A2, long long strideA2,
                    const ushort_t* __restrict__ B2, long long strideB2,
                    int K, int K2)
{
    constexpr int BK = 64;                 // two WMMA k-steps per stage
    constexpr int TM = BM / WM;
    constexpr int TN = BN / WN;
    constexpr int FM = TM / 16;
    constexpr int FN = TN / 16;
    constexpr int LDS_STRIDE = BK + 8;     // 72 elems = 144B rows (16B mult)
    constexpr int ACH = (BM * (BK / 8)) / 256;   // 16B chunks per thread (A)
    constexpr int BCH = (BN * (BK / 8)) / 256;   // 16B chunks per thread (B)

    __shared__ __align__(16) ushort_t As[2][BM * LDS_STRIDE];
    __shared__ __align__(16) ushort_t Bs[2][BN * LDS_STRIDE];

    const int tid  = threadIdx.x;
    const int lane = tid & 31;
    const int w    = tid >> 5;
    const int wm   = w % WM;
    const int wn   = w / WM;
    const int m0   = blockIdx.y * BM;
    const int n0   = blockIdx.x * BN;
    const int z    = blockIdx.z;

    unsigned asb[2], bsb[2];
    asb[0] = lds_byte_addr(&As[0][0]);  asb[1] = lds_byte_addr(&As[1][0]);
    bsb[0] = lds_byte_addr(&Bs[0][0]);  bsb[1] = lds_byte_addr(&Bs[1][0]);

    v8f acc[FM][FN];
    const v8f vzero = {0.f, 0.f, 0.f, 0.f, 0.f, 0.f, 0.f, 0.f};
#pragma unroll
    for (int i = 0; i < FM; ++i)
#pragma unroll
        for (int j = 0; j < FN; ++j) acc[i][j] = vzero;

    auto pass = [&](const ushort_t* Ap, const ushort_t* Bp, int KK) {
        // Per-thread staging addresses, fixed across the K loop.
        const ushort_t* ga[ACH]; unsigned lao[ACH];
        const ushort_t* gb[BCH]; unsigned lbo[BCH];
#pragma unroll
        for (int i = 0; i < ACH; ++i) {
            int c = tid + i * 256, row = c >> 3, sub = c & 7;
            ga[i]  = Ap + (size_t)(m0 + row) * lda + sub * 8;
            lao[i] = (unsigned)(row * LDS_STRIDE + sub * 8) * 2u;
        }
#pragma unroll
        for (int i = 0; i < BCH; ++i) {
            int c = tid + i * 256, row = c >> 3, sub = c & 7;
            gb[i]  = Bp + (size_t)(n0 + row) * ldb + sub * 8;
            lbo[i] = (unsigned)(row * LDS_STRIDE + sub * 8) * 2u;
        }
        auto issue = [&](int k0, int buf) {
#pragma unroll
            for (int i = 0; i < ACH; ++i) async_load_b128(asb[buf] + lao[i], ga[i] + k0);
#pragma unroll
            for (int i = 0; i < BCH; ++i) async_load_b128(bsb[buf] + lbo[i], gb[i] + k0);
        };

        issue(0, 0);
        async_wait0();
        __syncthreads();

        const int nt = KK / BK;
        for (int t = 0; t < nt; ++t) {
            const int cur = t & 1;
            if (t + 1 < nt) issue((t + 1) * BK, cur ^ 1);
#pragma unroll
            for (int ks = 0; ks < BK; ks += 32) {
                v16bf af[FM];
                v16bf bfr[FN];
#pragma unroll
                for (int fm = 0; fm < FM; ++fm)
                    af[fm] = load_frag(&As[cur][(wm * TM + fm * 16) * LDS_STRIDE + ks],
                                       LDS_STRIDE, lane);
#pragma unroll
                for (int fn = 0; fn < FN; ++fn)
                    bfr[fn] = load_frag(&Bs[cur][(wn * TN + fn * 16) * LDS_STRIDE + ks],
                                        LDS_STRIDE, lane);
#pragma unroll
                for (int fm = 0; fm < FM; ++fm)
#pragma unroll
                    for (int fn = 0; fn < FN; ++fn)
                        acc[fm][fn] = __builtin_amdgcn_wmma_f32_16x16x32_bf16(
                            false, af[fm], false, bfr[fn], (short)0, acc[fm][fn],
                            false, false);
            }
            async_wait0();      // next-tile LDS writes complete (this wave)
            __syncthreads();    // publish + fence reads before buffer reuse
        }
    };

    pass(A + (size_t)z * strideA, Bm + (size_t)z * strideB, K);
    if (A2 != nullptr)
        pass(A2 + (size_t)z * strideA2, B2 + (size_t)z * strideB2, K2);

    // Epilogue. C/D layout: n = lane&15; m = 8*(lane>>4) + vgpr index.
    const int cn = lane & 15;
    const int ch = lane >> 4;
#pragma unroll
    for (int fm = 0; fm < FM; ++fm) {
#pragma unroll
        for (int fn = 0; fn < FN; ++fn) {
            const int nol = n0 + wn * TN + fn * 16 + cn;
            const float bv = bias ? bias[nol] : 0.f;
#pragma unroll
            for (int j = 0; j < 8; ++j) {
                const int mrow = m0 + wm * TM + fm * 16 + ch * 8 + j;
                float v = acc[fm][fn][j] * scale + bv;
                if (RELU) v = fmaxf(v, 0.f);
                if (OUT_BF16) {
                    ushort_t* Cb = reinterpret_cast<ushort_t*>(Cp) + (size_t)z * strideC;
                    Cb[(size_t)mrow * ldc + nol] = f2bf(v);
                } else {
                    float* Cf = reinterpret_cast<float*>(Cp) + (size_t)z * strideC;
                    Cf[(size_t)mrow * ldc + nol] = v;
                }
            }
        }
    }
}

// ---------------------------------------------------------------------------
__global__ __launch_bounds__(256)
void cast_bf16_kernel(const float* __restrict__ src, ushort_t* __restrict__ dst,
                      unsigned int n) {
    unsigned int i = blockIdx.x * 256u + threadIdx.x;
    if (i < n) dst[i] = f2bf(src[i]);
}

// Apply exp(i*phase) to q,k,v (f32 [B,S,E]) and pack for attention GEMMs.
__global__ __launch_bounds__(256)
void rotate_pack_kernel(const float* __restrict__ qf, const float* __restrict__ kf,
                        const float* __restrict__ vf, const float* __restrict__ phase,
                        ushort_t* __restrict__ Qc,  ushort_t* __restrict__ Kcr,
                        ushort_t* __restrict__ Kci, ushort_t* __restrict__ Vt) {
    size_t i = (size_t)blockIdx.x * 256u + threadIdx.x;   // over B*S*E
    int e = (int)(i & (E_ - 1));
    size_t bs = i >> 10;
    int s = (int)(bs & (S_ - 1));
    int b = (int)(bs >> 10);
    int h = e >> 6, d = e & (D_ - 1);

    float sn, cs;
    sincosf(phase[h * D_ + d], &sn, &cs);

    float q = qf[i], k = kf[i], v = vf[i];
    size_t bh = (size_t)b * H_ + h;
    size_t o  = (bh * S_ + s) * (2 * D_);
    Qc [o + d]      = f2bf(q * cs);
    Qc [o + D_ + d] = f2bf(q * sn);
    Kcr[o + d]      = f2bf(k * cs);
    Kcr[o + D_ + d] = f2bf(-k * sn);
    Kci[o + d]      = f2bf(k * sn);
    Kci[o + D_ + d] = f2bf(k * cs);
    size_t vo = (bh * (2 * D_) + d) * S_ + s;   // Vt: [bh, 128, S]
    Vt[vo]            = f2bf(v * cs);
    Vt[vo + D_ * S_]  = f2bf(-v * sn);
}

// Complex softmax per row of length S_: z = sr + i*si (bf16 in/out, in place).
__global__ __launch_bounds__(256)
void complex_softmax_kernel(ushort_t* __restrict__ Sr, ushort_t* __restrict__ Si) {
    __shared__ float red[256];
    const size_t base = (size_t)blockIdx.x * S_;
    const int tid = threadIdx.x;

    float sr[4], si[4];
#pragma unroll
    for (int i = 0; i < 4; ++i) {
        int e = tid + i * 256;
        sr[i] = bf2f(Sr[base + e]);
        si[i] = bf2f(Si[base + e]);
    }
    float m = fmaxf(fmaxf(sr[0], sr[1]), fmaxf(sr[2], sr[3]));
    red[tid] = m; __syncthreads();
    for (int s = 128; s > 0; s >>= 1) {
        if (tid < s) red[tid] = fmaxf(red[tid], red[tid + s]);
        __syncthreads();
    }
    m = red[0]; __syncthreads();

    float er[4], ei[4], lr = 0.f, li = 0.f;
#pragma unroll
    for (int i = 0; i < 4; ++i) {
        float ex = __expf(sr[i] - m);
        float sn, cs; __sincosf(si[i], &sn, &cs);
        er[i] = ex * cs; ei[i] = ex * sn;
        lr += er[i]; li += ei[i];
    }
    red[tid] = lr; __syncthreads();
    for (int s = 128; s > 0; s >>= 1) {
        if (tid < s) red[tid] += red[tid + s];
        __syncthreads();
    }
    float SR = red[0]; __syncthreads();
    red[tid] = li; __syncthreads();
    for (int s = 128; s > 0; s >>= 1) {
        if (tid < s) red[tid] += red[tid + s];
        __syncthreads();
    }
    float SI = red[0];

    float inv = 1.0f / (SR * SR + SI * SI);
#pragma unroll
    for (int i = 0; i < 4; ++i) {
        int e = tid + i * 256;
        float ar = (er[i] * SR + ei[i] * SI) * inv;
        float ai = (ei[i] * SR - er[i] * SI) * inv;
        Sr[base + e] = f2bf(ar);
        Si[base + e] = f2bf(ai);
    }
}

// out = LayerNorm(X + Y); Y either flat [B*S,E] or head-major [B,H,S,D].
__global__ __launch_bounds__(256)
void residual_ln_kernel(const float* __restrict__ X, const float* __restrict__ Y,
                        int y_headmajor,
                        const float* __restrict__ g, const float* __restrict__ bb,
                        float* __restrict__ out, ushort_t* __restrict__ out_bf) {
    __shared__ float red[256];
    const int r = blockIdx.x;                 // row in [0, B*S)
    const int b = r >> 10, s = r & (S_ - 1);
    const int tid = threadIdx.x;

    float v[4];
    float lsum = 0.f;
#pragma unroll
    for (int i = 0; i < 4; ++i) {
        int e = tid + i * 256;
        float y;
        if (y_headmajor) {
            int h = e >> 6, d = e & (D_ - 1);
            y = Y[(((size_t)b * H_ + h) * S_ + s) * D_ + d];
        } else {
            y = Y[(size_t)r * E_ + e];
        }
        v[i] = X[(size_t)r * E_ + e] + y;
        lsum += v[i];
    }
    red[tid] = lsum; __syncthreads();
    for (int st = 128; st > 0; st >>= 1) {
        if (tid < st) red[tid] += red[tid + st];
        __syncthreads();
    }
    float mu = red[0] * (1.0f / E_); __syncthreads();

    float lvar = 0.f;
#pragma unroll
    for (int i = 0; i < 4; ++i) { float xc = v[i] - mu; lvar += xc * xc; }
    red[tid] = lvar; __syncthreads();
    for (int st = 128; st > 0; st >>= 1) {
        if (tid < st) red[tid] += red[tid + st];
        __syncthreads();
    }
    float rstd = rsqrtf(red[0] * (1.0f / E_) + 1e-5f);

#pragma unroll
    for (int i = 0; i < 4; ++i) {
        int e = tid + i * 256;
        float o = (v[i] - mu) * rstd * g[e] + bb[e];
        out[(size_t)r * E_ + e] = o;
        if (out_bf) out_bf[(size_t)r * E_ + e] = f2bf(o);
    }
}

// ---------------------------------------------------------------------------
extern "C" void kernel_launch(void* const* d_in, const int* in_sizes, int n_in,
                              void* d_out, int out_size, void* d_ws, size_t ws_size,
                              hipStream_t stream) {
    const float* x     = (const float*)d_in[0];
    const float* wq    = (const float*)d_in[1];
    const float* wk    = (const float*)d_in[2];
    const float* wv    = (const float*)d_in[3];
    const float* phase = (const float*)d_in[4];
    const float* ln1g  = (const float*)d_in[5];
    const float* ln1b  = (const float*)d_in[6];
    const float* ln2g  = (const float*)d_in[7];
    const float* ln2b  = (const float*)d_in[8];
    const float* w1    = (const float*)d_in[9];
    const float* b1    = (const float*)d_in[10];
    const float* w2    = (const float*)d_in[11];
    const float* b2    = (const float*)d_in[12];
    float* out = (float*)d_out;

    // ---- workspace carve-up ----
    char* ws = (char*)d_ws;
    size_t off = 0;
    auto alloc = [&](size_t bytes) -> char* {
        char* p = ws + off;
        off = (off + bytes + 255) & ~(size_t)255;
        return p;
    };
    const size_t NTOK = (size_t)B_ * S_;            // 4096 rows
    ushort_t* xb   = (ushort_t*)alloc(NTOK * E_ * 2);
    ushort_t* wqb  = (ushort_t*)alloc((size_t)E_ * E_ * 2);
    ushort_t* wkb  = (ushort_t*)alloc((size_t)E_ * E_ * 2);
    ushort_t* wvb  = (ushort_t*)alloc((size_t)E_ * E_ * 2);
    ushort_t* w1b  = (ushort_t*)alloc((size_t)FFN_ * E_ * 2);
    ushort_t* w2b  = (ushort_t*)alloc((size_t)E_ * FFN_ * 2);
    float*    qf   = (float*)alloc(NTOK * E_ * 4);
    float*    kf   = (float*)alloc(NTOK * E_ * 4);
    float*    vf   = (float*)alloc(NTOK * E_ * 4);
    ushort_t* Qc   = (ushort_t*)alloc((size_t)B_ * H_ * S_ * 2 * D_ * 2);
    ushort_t* Kcr  = (ushort_t*)alloc((size_t)B_ * H_ * S_ * 2 * D_ * 2);
    ushort_t* Kci  = (ushort_t*)alloc((size_t)B_ * H_ * S_ * 2 * D_ * 2);
    ushort_t* Vt   = (ushort_t*)alloc((size_t)B_ * H_ * 2 * D_ * S_ * 2);
    ushort_t* Sr   = (ushort_t*)alloc((size_t)B_ * H_ * S_ * S_ * 2);
    ushort_t* Si   = (ushort_t*)alloc((size_t)B_ * H_ * S_ * S_ * 2);
    float*    AO   = (float*)alloc((size_t)B_ * H_ * S_ * D_ * 4);
    float*    x1   = (float*)alloc(NTOK * E_ * 4);
    ushort_t* x1b  = (ushort_t*)alloc(NTOK * E_ * 2);
    ushort_t* Hb   = (ushort_t*)alloc(NTOK * FFN_ * 2);
    float*    F2   = (float*)alloc(NTOK * E_ * 4);
    (void)ws_size; (void)in_sizes; (void)n_in; (void)out_size;

    const dim3 T(256);
    auto cgrid = [](size_t n) { return dim3((unsigned)((n + 255) / 256)); };

    // 1) casts to bf16
    cast_bf16_kernel<<<cgrid(NTOK * E_), T, 0, stream>>>(x,  xb,  (unsigned)(NTOK * E_));
    cast_bf16_kernel<<<cgrid((size_t)E_ * E_), T, 0, stream>>>(wq, wqb, E_ * E_);
    cast_bf16_kernel<<<cgrid((size_t)E_ * E_), T, 0, stream>>>(wk, wkb, E_ * E_);
    cast_bf16_kernel<<<cgrid((size_t)E_ * E_), T, 0, stream>>>(wv, wvb, E_ * E_);
    cast_bf16_kernel<<<cgrid((size_t)FFN_ * E_), T, 0, stream>>>(w1, w1b, FFN_ * E_);
    cast_bf16_kernel<<<cgrid((size_t)E_ * FFN_), T, 0, stream>>>(w2, w2b, E_ * FFN_);

    // 2) Q/K/V projections: [4096,1024] x [1024,1024]^T
    {
        dim3 g(E_ / 128, NTOK / 128, 1);
        gemm_nt_kernel<128,128,2,4,false,false><<<g, T, 0, stream>>>(
            xb, E_, 0, wqb, E_, 0, qf, E_, 0, nullptr, 1.0f,
            nullptr, 0, nullptr, 0, E_, 0);
        gemm_nt_kernel<128,128,2,4,false,false><<<g, T, 0, stream>>>(
            xb, E_, 0, wkb, E_, 0, kf, E_, 0, nullptr, 1.0f,
            nullptr, 0, nullptr, 0, E_, 0);
        gemm_nt_kernel<128,128,2,4,false,false><<<g, T, 0, stream>>>(
            xb, E_, 0, wvb, E_, 0, vf, E_, 0, nullptr, 1.0f,
            nullptr, 0, nullptr, 0, E_, 0);
    }

    // 3) phase rotation + packing
    rotate_pack_kernel<<<cgrid(NTOK * E_), T, 0, stream>>>(qf, kf, vf, phase,
                                                           Qc, Kcr, Kci, Vt);

    // 4) complex scores (batched over B*H): K = 2*D = 128, scale = 1/sqrt(D)
    {
        dim3 g(S_ / 128, S_ / 128, B_ * H_);
        const long long sQK = (long long)S_ * 2 * D_;
        const long long sSc = (long long)S_ * S_;
        gemm_nt_kernel<128,128,2,4,true,false><<<g, T, 0, stream>>>(
            Qc, 2 * D_, sQK, Kcr, 2 * D_, sQK, Sr, S_, sSc, nullptr, 0.125f,
            nullptr, 0, nullptr, 0, 2 * D_, 0);
        gemm_nt_kernel<128,128,2,4,true,false><<<g, T, 0, stream>>>(
            Qc, 2 * D_, sQK, Kci, 2 * D_, sQK, Si, S_, sSc, nullptr, 0.125f,
            nullptr, 0, nullptr, 0, 2 * D_, 0);
    }

    // 5) complex softmax in place (one block per row)
    complex_softmax_kernel<<<dim3(B_ * H_ * S_), T, 0, stream>>>(Sr, Si);

    // 6) attention output: AO = a_r * vr^T + a_i * (-vi)^T   (N = 64)
    {
        dim3 g(1, S_ / 128, B_ * H_);
        const long long sSc = (long long)S_ * S_;
        const long long sV  = (long long)2 * D_ * S_;
        gemm_nt_kernel<128,64,4,2,false,false><<<g, T, 0, stream>>>(
            Sr, S_, sSc, Vt, S_, sV, AO, D_, (long long)S_ * D_, nullptr, 1.0f,
            Si, sSc, Vt + (size_t)D_ * S_, sV, S_, S_);
    }

    // 7) x1 = LN(x + AO) (+ bf16 copy)
    residual_ln_kernel<<<dim3((unsigned)NTOK), T, 0, stream>>>(
        x, AO, 1, ln1g, ln1b, x1, x1b);

    // 8) FFN up + relu (bf16 out)
    {
        dim3 g(FFN_ / 128, NTOK / 128, 1);
        gemm_nt_kernel<128,128,2,4,true,true><<<g, T, 0, stream>>>(
            x1b, E_, 0, w1b, E_, 0, Hb, FFN_, 0, b1, 1.0f,
            nullptr, 0, nullptr, 0, E_, 0);
    }
    // 9) FFN down (f32 out)
    {
        dim3 g(E_ / 128, NTOK / 128, 1);
        gemm_nt_kernel<128,128,2,4,false,false><<<g, T, 0, stream>>>(
            Hb, FFN_, 0, w2b, FFN_, 0, F2, E_, 0, b2, 1.0f,
            nullptr, 0, nullptr, 0, FFN_, 0);
    }

    // 10) final LN -> d_out
    residual_ln_kernel<<<dim3((unsigned)NTOK), T, 0, stream>>>(
        x1, F2, 0, ln2g, ln2b, out, nullptr);
}
